// MoELayer_45526653337966
// MI455X (gfx1250) — compile-verified, matrix operand order checked by reading, not run on hardware
//
#include <hip/hip_runtime.h>
#include <hip/hip_bf16.h>
#include <math.h>

// ---------------------------------------------------------------------------
// MoE layer for MI455X (gfx1250): top-2 grouped-GEMM FFN on
// v_wmma_f32_16x16x32_f16, weight staging via Tensor Data Mover (TDM).
// ---------------------------------------------------------------------------

typedef _Float16 v8h  __attribute__((ext_vector_type(8)));
typedef _Float16 v16h __attribute__((ext_vector_type(16)));
typedef float    v8f  __attribute__((ext_vector_type(8)));
typedef unsigned int u32x4 __attribute__((ext_vector_type(4)));
typedef int          i32x8 __attribute__((ext_vector_type(8)));
typedef int          i32x4 __attribute__((ext_vector_type(4)));

#define T_TOK  8192
#define H_DIM  1024
#define F_DIM  2048
#define E_NUM  8
#define TILE_M 32   // tokens per block
#define FCH    64   // F columns per chunk
#define KSUB   128  // stage-1 K sub-chunk
#define PSUB   128  // stage-2 output-column pass width

#if defined(__HIP_DEVICE_COMPILE__) && defined(__gfx1250__) && \
    __has_builtin(__builtin_amdgcn_tensor_load_to_lds)
#define USE_TDM 1
#else
#define USE_TDM 0
#endif

// Concatenate two 8-element f16 vectors into a WMMA A/B fragment (8 VGPRs).
__device__ __forceinline__ v16h cat16(v8h lo, v8h hi) {
  return __builtin_shufflevector(lo, hi, 0, 1, 2, 3, 4, 5, 6, 7,
                                          8, 9, 10, 11, 12, 13, 14, 15);
}

// Load one 16x32 (A) or 32x16 (B) f16 fragment where this lane's 32 K-values
// are contiguous at `p`. Per ISA layout: lanes 0-15 hold K=0..7,16..23,
// lanes 16-31 hold K=8..15,24..31  (koff = (lane>=16) ? 8 : 0).
__device__ __forceinline__ v16h frag_ld(const _Float16* p, int koff) {
  v8h lo = *(const v8h*)(p + koff);
  v8h hi = *(const v8h*)(p + 16 + koff);
  return cat16(lo, hi);
}

#if USE_TDM
// Issue a TDM 2D tile load: tile_h rows of tile_w contiguous f16 elements
// (global row stride `row_stride` elements) -> LDS, rows packed contiguously.
// D# layout per CDNA5 ISA 8.3/8.4 (data_size=1 -> 2-byte elements).
__device__ __forceinline__ void tdm_load_2d(unsigned lds_off,
                                            const _Float16* gptr, int tile_w,
                                            int tile_h, int row_stride) {
  const unsigned long long ga = (unsigned long long)(const void*)gptr;
  u32x4 g0;
  g0[0] = 1u;                                   // count=1 (valid), load, user
  g0[1] = lds_off;                              // lds_addr (bytes)
  g0[2] = (unsigned)(ga & 0xFFFFFFFFu);         // global_addr[31:0]
  g0[3] = (unsigned)((ga >> 32) & 0x1FFFFFFu)   // global_addr[56:32]
          | (2u << 30);                         // type=2 ("image")
  i32x8 g1;
  g1[0] = 1 << 16;                              // data_size=1 (2B)
  g1[1] = (tile_w & 0xFFFF) << 16;              // tensor_dim0[15:0]
  g1[2] = ((unsigned)tile_w >> 16) |            // tensor_dim0[31:16]
          ((tile_h & 0xFFFF) << 16);            // tensor_dim1[15:0]
  g1[3] = (tile_w & 0xFFFF) << 16;              // tensor_dim1[31:16]=0 | tile_dim0
  g1[4] = tile_h & 0xFFFF;                      // tile_dim1 | tile_dim2=0
  g1[5] = row_stride;                           // tensor_dim0_stride[31:0]
  g1[6] = 0;                                    // stride0[47:32] | stride1[15:0]
  g1[7] = 0;                                    // stride1[47:16]
  const i32x4 gz4 = {0, 0, 0, 0};               // 2D: groups 2/3 unused
  const i32x8 gz8 = {0, 0, 0, 0, 0, 0, 0, 0};
  __builtin_amdgcn_tensor_load_to_lds(g0, g1, gz4, gz4, gz8, 0);
}
#endif

// ---------------------------------------------------------------------------
// Workspace init + output zero
// ---------------------------------------------------------------------------
__global__ void init_ws_kernel(int* counts, float* blkAux, float* blkZ) {
  int i = threadIdx.x;
  if (i < E_NUM) counts[i] = 0;
  if (i < 32) { blkAux[i] = 0.f; blkZ[i] = 0.f; }
}

__global__ void zero_out_kernel(float* out, int n) {
  int i = blockIdx.x * blockDim.x + threadIdx.x;
  if (i < n) out[i] = 0.f;
}

// ---------------------------------------------------------------------------
// fp32 -> f16 conversion of activations
// ---------------------------------------------------------------------------
__global__ void cvt_x_kernel(const float* __restrict__ src,
                             _Float16* __restrict__ dst, int n) {
  int i = blockIdx.x * blockDim.x + threadIdx.x;
  if (i < n) dst[i] = (_Float16)src[i];
}

// ---------------------------------------------------------------------------
// Per-expert transpose + fp32 -> f16: dst[e][c][r] = (f16) src[e][r][c]
// ---------------------------------------------------------------------------
__global__ void transpose_cvt_kernel(const float* __restrict__ src,
                                     _Float16* __restrict__ dst,
                                     int R, int C) {
  __shared__ _Float16 tile[32][33];
  const int e  = blockIdx.z;
  const size_t base = (size_t)e * R * C;
  const int c0 = blockIdx.x * 32;
  const int r0 = blockIdx.y * 32;
  const int tx = threadIdx.x, ty = threadIdx.y;
  for (int i = ty; i < 32; i += 8)
    tile[i][tx] = (_Float16)src[base + (size_t)(r0 + i) * C + (c0 + tx)];
  __syncthreads();
  for (int i = ty; i < 32; i += 8)
    dst[base + (size_t)(c0 + i) * R + (r0 + tx)] = tile[tx][i];
}

// ---------------------------------------------------------------------------
// Router: exact fp32 logits, log-softmax, top-2 routing lists, loss partials.
// ---------------------------------------------------------------------------
__global__ __launch_bounds__(256) void router_kernel(
    const float* __restrict__ x, const float* __restrict__ rw,
    int* __restrict__ counts, int* __restrict__ lists,
    float* __restrict__ blkAux, float* __restrict__ blkZ) {
  __shared__ float rws[E_NUM * H_DIM];
  __shared__ float redA[8], redZ[8];
  const int tid = threadIdx.x;
  for (int i = tid; i < E_NUM * H_DIM; i += 256) rws[i] = rw[i];
  __syncthreads();

  const int t = blockIdx.x * 256 + tid;
  const float4* xr = (const float4*)(x + (size_t)t * H_DIM);
  float acc[8] = {0.f, 0.f, 0.f, 0.f, 0.f, 0.f, 0.f, 0.f};
  for (int k = 0; k < H_DIM / 4; ++k) {
    const float4 xv = xr[k];
#pragma unroll
    for (int e2 = 0; e2 < 8; ++e2) {
      const float4 wv = *(const float4*)(rws + e2 * H_DIM + k * 4);
      acc[e2] += xv.x * wv.x + xv.y * wv.y + xv.z * wv.z + xv.w * wv.w;
    }
  }

  // top-2 (ties -> lowest index, matching lax.top_k)
  float m = acc[0]; int bi = 0;
#pragma unroll
  for (int e2 = 1; e2 < 8; ++e2) if (acc[e2] > m) { m = acc[e2]; bi = e2; }
  float s2 = -3.4e38f; int si = 0;
#pragma unroll
  for (int e2 = 0; e2 < 8; ++e2)
    if (e2 != bi && acc[e2] > s2) { s2 = acc[e2]; si = e2; }

  float se = 0.f, suml = 0.f, sumsq = 0.f;
#pragma unroll
  for (int e2 = 0; e2 < 8; ++e2) {
    se += expf(acc[e2] - m);
    suml += acc[e2];
    sumsq += acc[e2] * acc[e2];
  }
  const float lse = m + logf(se);
  const float LOGI = -2.0794415416798357f;  // log(1/8)
  const float a_t = 8.f * LOGI - (suml - 8.f * lse);

  int p1 = atomicAdd(&counts[bi], 1); lists[bi * T_TOK + p1] = t;
  int p2 = atomicAdd(&counts[si], 1); lists[si * T_TOK + p2] = t;

  float a = a_t, z = sumsq;
  for (int off = 16; off > 0; off >>= 1) {
    a += __shfl_down(a, off, 32);
    z += __shfl_down(z, off, 32);
  }
  if ((tid & 31) == 0) { redA[tid >> 5] = a; redZ[tid >> 5] = z; }
  __syncthreads();
  if (tid == 0) {
    float A = 0.f, Z = 0.f;
    for (int w = 0; w < 8; ++w) { A += redA[w]; Z += redZ[w]; }
    blkAux[blockIdx.x] = A;
    blkZ[blockIdx.x] = Z;
  }
}

__global__ void finalize_kernel(const float* __restrict__ blkAux,
                                const float* __restrict__ blkZ,
                                float* __restrict__ outLoss) {
  if (threadIdx.x == 0 && blockIdx.x == 0) {
    float A = 0.f, Z = 0.f;
    for (int i = 0; i < 32; ++i) { A += blkAux[i]; Z += blkZ[i]; }
    const float aux = 0.125f * A / 4.0f * 0.001f;
    const float zl  = Z / (float)(T_TOK * E_NUM) * 0.001f;
    *outLoss = aux + zl;
  }
}

// ---------------------------------------------------------------------------
// Fused grouped-GEMM FFN per (expert, 32-token tile), 8 waves.
//   stage-1: H[32x64] = silu(X[32x1024] @ W1 + b1); W1 sub-chunks via TDM.
//   stage-2: Y[32x1024] += H[32x64] @ W2; W2 pass tiles via TDM.
//   16 v8f accumulators/wave (128 VGPRs).
// ---------------------------------------------------------------------------
__global__ __launch_bounds__(256, 1) void ffn_kernel(
    const _Float16* __restrict__ Xh,   // [T][H] f16
    const _Float16* __restrict__ W1t,  // [E][F][H] f16 (transposed)
    const _Float16* __restrict__ W2t,  // [E][H][F] f16 (transposed)
    const float* __restrict__ b1,      // [E][F]
    const float* __restrict__ b2,      // [E][H]
    const int* __restrict__ counts,
    const int* __restrict__ lists,     // [E][T]
    float* __restrict__ out) {         // [T][H] (pre-zeroed)
  const int e = blockIdx.y;
  const int n = counts[e];
  const int tile0 = blockIdx.x * TILE_M;
  if (tile0 >= n) return;

  const int tid  = threadIdx.x;
  const int wave = tid >> 5;
  const int lane = tid & 31;
  const int hl   = lane >> 4;
  const int ln   = lane & 15;
  const int koff = hl * 8;

  const int rw = wave & 1;   // row tile (16 tokens)
  const int cq = wave >> 1;  // column group (0..3)

  __shared__ _Float16 Xs[TILE_M * KSUB];   // [row][k']      8 KB
  __shared__ _Float16 W1s[FCH * KSUB];     // [f'][k']      16 KB
  __shared__ _Float16 W2s[PSUB * FCH];     // [h'][k']      16 KB
  __shared__ _Float16 Hs[TILE_M * FCH];    // [row][f']      4 KB
  __shared__ int tokS[TILE_M];

  if (tid < TILE_M) {
    const int r = tile0 + tid;
    tokS[tid] = (r < n) ? lists[e * T_TOK + r] : lists[e * T_TOK + tile0];
  }

  v8f yacc[16];
  const v8f vzero = {0.f, 0.f, 0.f, 0.f, 0.f, 0.f, 0.f, 0.f};
#pragma unroll
  for (int i = 0; i < 16; ++i) yacc[i] = vzero;

  for (int fc = 0; fc < F_DIM / FCH; ++fc) {
    const int fbase = fc * FCH;
    v8f hacc = vzero;

    // ---- stage 1: K = 1024 in 8 sub-chunks of 128 ----
    for (int ks = 0; ks < H_DIM / KSUB; ++ks) {
      __syncthreads();
#if USE_TDM
      if (wave == 0) {  // TDM: 64 rows x 128 f16, row stride H
        tdm_load_2d((unsigned)(size_t)W1s,
                    W1t + ((size_t)(e * F_DIM + fbase)) * H_DIM + ks * KSUB,
                    KSUB, FCH, H_DIM);
      }
#endif
      {
        // Xs gather: 32x128 f16 = 512 x 16B chunks (2 per thread)
        float4 tx[2];
#pragma unroll
        for (int it = 0; it < 2; ++it) {
          const int i = tid + it * 256;
          const int r = i >> 4, q = i & 15;
          tx[it] = *(const float4*)(Xh + (size_t)tokS[r] * H_DIM + ks * KSUB +
                                    q * 8);
        }
#if !USE_TDM
        float4 tw[4];
#pragma unroll
        for (int it = 0; it < 4; ++it) {
          const int i = tid + it * 256;
          const int f = i >> 4, q = i & 15;
          tw[it] = *(const float4*)(W1t +
                                    ((size_t)(e * F_DIM + fbase + f)) * H_DIM +
                                    ks * KSUB + q * 8);
        }
#endif
#pragma unroll
        for (int it = 0; it < 2; ++it) {
          const int i = tid + it * 256;
          *(float4*)(Xs + (i >> 4) * KSUB + (i & 15) * 8) = tx[it];
        }
#if !USE_TDM
#pragma unroll
        for (int it = 0; it < 4; ++it) {
          const int i = tid + it * 256;
          *(float4*)(W1s + (i >> 4) * KSUB + (i & 15) * 8) = tw[it];
        }
#endif
      }
#if USE_TDM
      if (wave == 0) __builtin_amdgcn_s_wait_tensorcnt(0);
#endif
      __syncthreads();
#pragma unroll
      for (int kk = 0; kk < KSUB / 32; ++kk) {
        const v16h a = frag_ld(Xs + (rw * 16 + ln) * KSUB + kk * 32, koff);
        const v16h b = frag_ld(W1s + (cq * 16 + ln) * KSUB + kk * 32, koff);
        hacc = __builtin_amdgcn_wmma_f32_16x16x32_f16(
            false, a, false, b, (short)0, hacc, false, false);
      }
    }

    // ---- bias + silu -> Hs ----
    {
      const int col = cq * 16 + ln;
      const float bb = b1[e * F_DIM + fbase + col];
#pragma unroll
      for (int v = 0; v < 8; ++v) {
        const float xv = hacc[v] + bb;
        const float sv = xv / (1.0f + __expf(-xv));
        Hs[(rw * 16 + hl * 8 + v) * FCH + col] = (_Float16)sv;
      }
    }

    // ---- stage 2: 8 passes of 128 output columns ----
    for (int p = 0; p < H_DIM / PSUB; ++p) {
      __syncthreads();
#if USE_TDM
      if (wave == 0) {  // TDM: 128 rows x 64 f16, row stride F
        tdm_load_2d((unsigned)(size_t)W2s,
                    W2t + ((size_t)(e * H_DIM + p * PSUB)) * F_DIM + fbase,
                    FCH, PSUB, F_DIM);
        __builtin_amdgcn_s_wait_tensorcnt(0);
      }
#else
      {
        float4 tw[4];
#pragma unroll
        for (int it = 0; it < 4; ++it) {
          const int i = tid + it * 256;
          const int hp = i >> 3, q = i & 7;
          tw[it] = *(const float4*)(W2t +
                                    ((size_t)(e * H_DIM + p * PSUB + hp)) *
                                        F_DIM +
                                    fbase + q * 8);
        }
#pragma unroll
        for (int it = 0; it < 4; ++it) {
          const int i = tid + it * 256;
          *(float4*)(W2s + (i >> 3) * FCH + (i & 7) * 8) = tw[it];
        }
      }
#endif
      __syncthreads();
#pragma unroll
      for (int kst = 0; kst < FCH / 32; ++kst) {
        const v16h a2 = frag_ld(Hs + (rw * 16 + ln) * FCH + kst * 32, koff);
#pragma unroll
        for (int j = 0; j < 2; ++j) {
          const int lt = cq * 2 + j;  // local col tile within pass (0..7)
          const v16h b = frag_ld(W2s + (lt * 16 + ln) * FCH + kst * 32, koff);
          yacc[p * 2 + j] = __builtin_amdgcn_wmma_f32_16x16x32_f16(
              false, a2, false, b, (short)0, yacc[p * 2 + j], false, false);
        }
      }
    }
  }

  // ---- epilogue: + b2, scatter-add into out (2 adds/element total) ----
  int toks[8];
  bool val[8];
#pragma unroll
  for (int v = 0; v < 8; ++v) {
    const int r = rw * 16 + hl * 8 + v;
    val[v] = (tile0 + r < n);
    toks[v] = tokS[r];
  }
#pragma unroll
  for (int p = 0; p < 8; ++p) {
#pragma unroll
    for (int j = 0; j < 2; ++j) {
      const int col = (p * 8 + cq * 2 + j) * 16 + ln;
      const float bb = b2[e * H_DIM + col];
      const v8f y = yacc[p * 2 + j];
#pragma unroll
      for (int v = 0; v < 8; ++v) {
        if (val[v]) atomicAdd(out + (size_t)toks[v] * H_DIM + col, y[v] + bb);
      }
    }
  }
}

// ---------------------------------------------------------------------------
// Host launcher
// ---------------------------------------------------------------------------
extern "C" void kernel_launch(void* const* d_in, const int* in_sizes, int n_in,
                              void* d_out, int out_size, void* d_ws,
                              size_t ws_size, hipStream_t stream) {
  const float* x  = (const float*)d_in[0];
  const float* rw = (const float*)d_in[1];
  const float* w1 = (const float*)d_in[2];
  const float* b1 = (const float*)d_in[3];
  const float* w2 = (const float*)d_in[4];
  const float* b2 = (const float*)d_in[5];
  float* out = (float*)d_out;

  char* ws = (char*)d_ws;
  int*   counts = (int*)(ws + 0);
  float* blkAux = (float*)(ws + 256);
  float* blkZ   = (float*)(ws + 512);
  int*   lists  = (int*)(ws + 1024);                 // 8*8192*4 = 256KB
  _Float16* Xh  = (_Float16*)(ws + (1ull  << 20));   // 16 MiB
  _Float16* W1t = (_Float16*)(ws + (18ull << 20));   // 32 MiB
  _Float16* W2t = (_Float16*)(ws + (52ull << 20));   // 32 MiB  (total ~84 MiB)

  const int out_elems = T_TOK * H_DIM + 1;

  init_ws_kernel<<<1, 64, 0, stream>>>(counts, blkAux, blkZ);
  zero_out_kernel<<<(out_elems + 255) / 256, 256, 0, stream>>>(out, out_elems);
  cvt_x_kernel<<<(T_TOK * H_DIM) / 256, 256, 0, stream>>>(x, Xh, T_TOK * H_DIM);
  transpose_cvt_kernel<<<dim3(F_DIM / 32, H_DIM / 32, E_NUM), dim3(32, 8), 0,
                         stream>>>(w1, W1t, H_DIM, F_DIM);
  transpose_cvt_kernel<<<dim3(H_DIM / 32, F_DIM / 32, E_NUM), dim3(32, 8), 0,
                         stream>>>(w2, W2t, F_DIM, H_DIM);
  router_kernel<<<T_TOK / 256, 256, 0, stream>>>(x, rw, counts, lists, blkAux,
                                                 blkZ);
  ffn_kernel<<<dim3(T_TOK / TILE_M, E_NUM), 256, 0, stream>>>(
      Xh, W1t, W2t, b1, b2, counts, lists, out);
  finalize_kernel<<<1, 32, 0, stream>>>(blkAux, blkZ, out + T_TOK * H_DIM);
}
